// HierarchicalTransformer_17643725652008
// MI455X (gfx1250) — matrix operand order repaired and checked by reading, not compiled, not run on hardware
//
#include <hip/hip_runtime.h>

// ---------------------------------------------------------------------------
// Longformer-style encoder (B=1, S=4096, H=12, D=64, E=768, W=256, G=64, L=12)
// for MI455X / gfx1250.
//   * All GEMM / attention matmul math on v_wmma_f32_16x16x32_f16 (f32 accum).
//   * Per-layer weights transposed+converted once to f16 [N,K] (17.7MB arena,
//     reused across layers), so GEMM B tiles stream global->LDS via the
//     gfx1250 async-to-LDS path (ASYNCcnt) in exactly the fragment-friendly
//     [col][k] layout. Double-buffered LDS + register-prefetched A tiles.
//   * Workload is compute bound: ~1 TFLOP of matmul vs 425MB of weights
//     (weights L2-resident per layer; activations 12.6MB).
// Workspace use: ~157 MB.
// ---------------------------------------------------------------------------

constexpr int SEQ = 4096, EMB = 768, NH = 12, HD = 64, WIN = 256, GLB = 64,
              NL = 12, FFD = 3072, VOCAB = 50282;
constexpr float NEGF = -1.0e9f;

typedef __attribute__((ext_vector_type(16))) _Float16 v16h;
typedef __attribute__((ext_vector_type(8)))  float    v8f;

#define DEVI __device__ __forceinline__

#ifndef __has_builtin
#define __has_builtin(x) 0
#endif
#if __has_builtin(__builtin_amdgcn_global_load_async_to_lds_b128)
#define HAVE_ASYNC_LDS 1
#else
#define HAVE_ASYNC_LDS 0
#endif

union F16Frag { v16h v; _Float16 h[16]; unsigned u[8]; };

DEVI v8f wmma32(v16h a, v16h b, v8f c) {
  // (neg_a, A, neg_b, B, c_mod, C, reuse_a, reuse_b)
  return __builtin_amdgcn_wmma_f32_16x16x32_f16(false, a, false, b, (short)0, c,
                                                false, false);
}

DEVI v8f v8f_zero() { v8f z = {0.f,0.f,0.f,0.f,0.f,0.f,0.f,0.f}; return z; }

#if HAVE_ASYNC_LDS
typedef __attribute__((vector_size(16))) int async_v4i;
DEVI void async_b128(const _Float16* g, _Float16* l) {
  typedef __attribute__((address_space(1))) async_v4i* gas1;
  typedef __attribute__((address_space(3))) async_v4i* las3;
  __builtin_amdgcn_global_load_async_to_lds_b128(
      (gas1)(unsigned long long)g,
      (las3)(unsigned)(unsigned long long)l, 0, 0);
}
#endif

DEVI void wait_async() {
#if HAVE_ASYNC_LDS
#if __has_builtin(__builtin_amdgcn_s_wait_asynccnt)
  __builtin_amdgcn_s_wait_asynccnt(0);
#else
  asm volatile("s_wait_asynccnt 0" ::: "memory");
#endif
#endif
}

// A-fragment (16x32 f16) from LDS, rows = lane%16, contiguous K order:
// lane-half 0 -> K {0..15}, lane-half 1 -> K {16..31}
DEVI v16h load_a_frag(const _Float16* __restrict__ base, int rowBase, int lane,
                      int stride) {
  F16Frag f;
  const int row = rowBase + (lane & 15);
  const int kb = (lane >> 4) * 16;
  const _Float16* p = base + row * stride;
#pragma unroll
  for (int v = 0; v < 8; ++v) {
    const int k0 = kb + 2 * v;
    f.u[v] = *(const unsigned*)(p + k0);  // 2 consecutive halves, 4B aligned
  }
  return f.v;
}

// B-fragment (32x16 f16) from LDS stored N-major [col][k]:
// col = lane%16, K = (lane/16)*16 + 2v, 2v+1
DEVI v16h load_b_frag(const _Float16* __restrict__ base, int colBase, int lane,
                      int stride) {
  F16Frag f;
  const int col = colBase + (lane & 15);
  const int kb = (lane >> 4) * 16;
  const _Float16* p = base + col * stride + kb;
#pragma unroll
  for (int v = 0; v < 8; ++v) f.u[v] = *(const unsigned*)(p + 2 * v);
  return f.v;
}

// ---------------------------------------------------------------------------
// Weight transpose + f32->f16 convert: src[K,N] -> dst[N,K]. block (32,8)
// ---------------------------------------------------------------------------
__global__ __launch_bounds__(256)
void transpose_f16_kernel(const float* __restrict__ src,
                          _Float16* __restrict__ dst, int K, int N) {
  __shared__ _Float16 t[32][33];
  const int n0 = blockIdx.x * 32, k0 = blockIdx.y * 32;
#pragma unroll
  for (int j = 0; j < 4; ++j) {
    const int k = k0 + threadIdx.y + 8 * j;
    t[threadIdx.y + 8 * j][threadIdx.x] =
        (_Float16)src[(size_t)k * N + n0 + threadIdx.x];
  }
  __syncthreads();
#pragma unroll
  for (int j = 0; j < 4; ++j) {
    const int n = n0 + threadIdx.y + 8 * j;
    dst[(size_t)n * K + k0 + threadIdx.x] = t[threadIdx.x][threadIdx.y + 8 * j];
  }
}

// ---------------------------------------------------------------------------
// GEMM: C[M,N] = (A[M,K] @ B[K,N] + bias[N]) * post_scale
//   A fp32 row-major, B supplied TRANSPOSED fp16 [N,K] row-major.
// Block 256 thr = 8 waves (2x4), block tile 64x256, wave tile 32x64, BK=64.
// Double-buffered LDS; B tiles via async global->LDS (fallback: uint4 copy);
// A tiles register-prefetched and converted f32->f16 in the shadow of WMMA.
// ---------------------------------------------------------------------------
__global__ __launch_bounds__(256)
void gemm_bt_kernel(const float* __restrict__ A, const _Float16* __restrict__ BT,
                    const float* __restrict__ bias, float* __restrict__ C,
                    int M, int N, int K, float post_scale) {
  __shared__ _Float16 As[2][64 * 72];   // [row][k0..63], stride 72 halves
  __shared__ _Float16 Bs[2][256 * 72];  // [col][k0..63], stride 72 halves
  const int tid = threadIdx.x;
  const int lane = tid & 31, wave = tid >> 5;
  const int wm = wave >> 2, wn = wave & 3;
  const int bm = blockIdx.y * 64, bn = blockIdx.x * 256;
  const int lh = lane & 15, lhi = lane >> 4;
  (void)M;

  v8f acc[2][4];
#pragma unroll
  for (int i = 0; i < 2; ++i)
#pragma unroll
    for (int j = 0; j < 4; ++j) acc[i][j] = v8f_zero();

  const int ar = tid >> 2, ac = (tid & 3) * 16;  // A: 16 floats / thread
  const int bcol = tid;                          // B: one col / thread, 8xb128

  auto issueB = [&](int k0, int buf) {
    const _Float16* g = BT + (size_t)(bn + bcol) * K + k0;
    _Float16* l = &Bs[buf][bcol * 72];
#pragma unroll
    for (int kg = 0; kg < 8; ++kg) {
#if HAVE_ASYNC_LDS
      async_b128(g + kg * 8, l + kg * 8);
#else
      *(uint4*)(l + kg * 8) = *(const uint4*)(g + kg * 8);
#endif
    }
  };
  auto loadA = [&](int k0, float4 (&r)[4]) {
    const float* sa = A + (size_t)(bm + ar) * K + k0 + ac;
#pragma unroll
    for (int j = 0; j < 4; ++j) r[j] = *(const float4*)(sa + 4 * j);
  };
  auto storeA = [&](int buf, const float4 (&r)[4]) {
#pragma unroll
    for (int j = 0; j < 4; ++j) {
      union { _Float16 h[4]; uint2 u; } p;
      p.h[0] = (_Float16)r[j].x; p.h[1] = (_Float16)r[j].y;
      p.h[2] = (_Float16)r[j].z; p.h[3] = (_Float16)r[j].w;
      *(uint2*)&As[buf][ar * 72 + ac + 4 * j] = p.u;
    }
  };

  float4 areg[4];
  issueB(0, 0);
  loadA(0, areg);
  storeA(0, areg);
  wait_async();
  __syncthreads();

  int cur = 0;
  for (int k0 = 0; k0 < K; k0 += 64) {
    const int nxt = k0 + 64;
    const bool more = nxt < K;
    if (more) {
      issueB(nxt, cur ^ 1);  // async into idle buffer while we compute
      loadA(nxt, areg);      // prefetch A into registers
    }
#pragma unroll
    for (int ks = 0; ks < 2; ++ks) {
      const _Float16* ab = &As[cur][ks * 32];
      const _Float16* bb = &Bs[cur][ks * 32];
      const v16h a0 = load_a_frag(ab, wm * 32, lane, 72);
      const v16h a1 = load_a_frag(ab, wm * 32 + 16, lane, 72);
#pragma unroll
      for (int ni = 0; ni < 4; ++ni) {
        const v16h bf = load_b_frag(bb, wn * 64 + ni * 16, lane, 72);
        acc[0][ni] = wmma32(a0, bf, acc[0][ni]);
        acc[1][ni] = wmma32(a1, bf, acc[1][ni]);
      }
    }
    if (more) storeA(cur ^ 1, areg);
    wait_async();
    __syncthreads();
    cur ^= 1;
  }
#pragma unroll
  for (int mi = 0; mi < 2; ++mi)
#pragma unroll
    for (int ni = 0; ni < 4; ++ni) {
      const int col = bn + wn * 64 + ni * 16 + lh;
      const float bval = bias[col];
#pragma unroll
      for (int r = 0; r < 8; ++r) {
        const int row = bm + wm * 32 + mi * 16 + r + 8 * lhi;
        C[(size_t)row * N + col] = (acc[mi][ni][r] + bval) * post_scale;
      }
    }
}

// ---------------------------------------------------------------------------
// Attention chunk: 32 queries x 32 keys, flash-style online softmax.
// ---------------------------------------------------------------------------
template <typename KeyFn, typename AddFn>
DEVI void attn_chunk(const float* __restrict__ Kp, const float* __restrict__ Vp,
                     int co, int lane, const v16h (&qa)[2][2], v8f (&o)[2][4],
                     float (&m_st)[2][8], float (&l_st)[2][8],
                     _Float16* __restrict__ plds, KeyFn keyf, AddFn addf) {
  const int lh = lane & 15, lhi = lane >> 4;
  // K fragments (B layout): contraction = d (2 WMMA k-chunks), N = 32 keys
  v16h kf[2][2];
#pragma unroll
  for (int ni = 0; ni < 2; ++ni) {
    const int key = keyf(ni * 16 + lh);
    const float* kr = Kp + (size_t)key * EMB + co + lhi * 16;
#pragma unroll
    for (int kk = 0; kk < 2; ++kk) {
      F16Frag f;
#pragma unroll
      for (int vv = 0; vv < 8; ++vv) {
        float2 t = *(const float2*)(kr + kk * 32 + 2 * vv);
        f.h[2 * vv] = (_Float16)t.x;
        f.h[2 * vv + 1] = (_Float16)t.y;
      }
      kf[kk][ni] = f.v;
    }
  }
  // scores = Q K^T
  v8f sc[2][2];
#pragma unroll
  for (int mi = 0; mi < 2; ++mi)
#pragma unroll
    for (int ni = 0; ni < 2; ++ni) {
      v8f z = v8f_zero();
      z = wmma32(qa[mi][0], kf[0][ni], z);
      z = wmma32(qa[mi][1], kf[1][ni], z);
      sc[mi][ni] = z;
    }
  // masks + online softmax (row = mi*16 + r + 8*(lane/16), col = ni*16+lane%16)
#pragma unroll
  for (int mi = 0; mi < 2; ++mi) {
    float rmax[8], rsum[8];
#pragma unroll
    for (int r = 0; r < 8; ++r) {
      const int qrow = mi * 16 + r + 8 * lhi;
      sc[mi][0][r] += addf(qrow, lh);
      sc[mi][1][r] += addf(qrow, 16 + lh);
      rmax[r] = fmaxf(sc[mi][0][r], sc[mi][1][r]);
    }
#pragma unroll
    for (int r = 0; r < 8; ++r)
#pragma unroll
      for (int s = 1; s < 16; s <<= 1)
        rmax[r] = fmaxf(rmax[r], __shfl_xor(rmax[r], s, 32));
#pragma unroll
    for (int r = 0; r < 8; ++r) {
      const float nm = fmaxf(m_st[mi][r], rmax[r]);
      const float sf = __expf(m_st[mi][r] - nm);
      m_st[mi][r] = nm;
      l_st[mi][r] *= sf;
      const float p0 = __expf(sc[mi][0][r] - nm);
      const float p1 = __expf(sc[mi][1][r] - nm);
      sc[mi][0][r] = p0;
      sc[mi][1][r] = p1;
      rsum[r] = p0 + p1;
#pragma unroll
      for (int ni = 0; ni < 4; ++ni) o[mi][ni][r] *= sf;
    }
#pragma unroll
    for (int r = 0; r < 8; ++r) {
#pragma unroll
      for (int s = 1; s < 16; s <<= 1) rsum[r] += __shfl_xor(rsum[r], s, 32);
      l_st[mi][r] += rsum[r];
    }
  }
  // P (C-layout, f32) -> LDS f16, then reload as A-frags (layout transpose).
  // Same-wave LDS is in-order in HW; only a compiler fence is needed.
#pragma unroll
  for (int mi = 0; mi < 2; ++mi)
#pragma unroll
    for (int ni = 0; ni < 2; ++ni)
#pragma unroll
      for (int r = 0; r < 8; ++r)
        plds[(mi * 16 + r + 8 * lhi) * 34 + ni * 16 + lh] =
            (_Float16)sc[mi][ni][r];
  asm volatile("" ::: "memory");
  v16h pa[2];
  pa[0] = load_a_frag(plds, 0, lane, 34);
  pa[1] = load_a_frag(plds, 16, lane, 34);
  // V fragments (B layout): contraction = 32 keys, N = 64 d
  v16h vf[4];
#pragma unroll
  for (int ni = 0; ni < 4; ++ni) {
    const int dcol = co + ni * 16 + lh;
    F16Frag f;
#pragma unroll
    for (int vv = 0; vv < 8; ++vv) {
      const int c0 = lhi * 16 + 2 * vv;
      f.h[2 * vv] = (_Float16)Vp[(size_t)keyf(c0) * EMB + dcol];
      f.h[2 * vv + 1] = (_Float16)Vp[(size_t)keyf(c0 + 1) * EMB + dcol];
    }
    vf[ni] = f.v;
  }
  asm volatile("" ::: "memory");
#pragma unroll
  for (int mi = 0; mi < 2; ++mi)
#pragma unroll
    for (int ni = 0; ni < 4; ++ni) o[mi][ni] = wmma32(pa[mi], vf[ni], o[mi][ni]);
}

DEVI v16h load_q_frag(const float* __restrict__ Qp, int rowBase, int co, int kk,
                      int lane) {
  const int lh = lane & 15, kb = (lane >> 4) * 16;
  const float* qr = Qp + (size_t)(rowBase + lh) * EMB + co;
  F16Frag f;
#pragma unroll
  for (int vv = 0; vv < 8; ++vv) {
    const int d = kk * 32 + kb + 2 * vv;
    float2 t = *(const float2*)(qr + d);
    f.h[2 * vv] = (_Float16)t.x;
    f.h[2 * vv + 1] = (_Float16)t.y;
  }
  return f.v;
}

// ---------------------------------------------------------------------------
// Local (windowed + global-key) attention: grid (S/W, H), 8 waves x 32 queries
// ---------------------------------------------------------------------------
__global__ __launch_bounds__(256)
void local_attn_kernel(const float* __restrict__ Qp, const float* __restrict__ Kp,
                       const float* __restrict__ Vp,
                       const float* __restrict__ madd,
                       const float* __restrict__ wrem, float* __restrict__ Op) {
  __shared__ _Float16 plds_all[8 * 32 * 34];
  const int tid = threadIdx.x, lane = tid & 31, wave = tid >> 5;
  _Float16* plds = plds_all + wave * (32 * 34);
  const int blk = blockIdx.x, head = blockIdx.y;
  const int co = head * HD;
  const int qseq = blk * WIN + wave * 32;  // first query row of this wave
  const int qinb = wave * 32;              // query offset within window block
  const int lh = lane & 15, lhi = lane >> 4;

  v16h qa[2][2];
#pragma unroll
  for (int mi = 0; mi < 2; ++mi)
#pragma unroll
    for (int kk = 0; kk < 2; ++kk)
      qa[mi][kk] = load_q_frag(Qp, qseq + mi * 16, co, kk, lane);

  v8f o[2][4];
  float m_st[2][8], l_st[2][8];
#pragma unroll
  for (int mi = 0; mi < 2; ++mi) {
#pragma unroll
    for (int ni = 0; ni < 4; ++ni) o[mi][ni] = v8f_zero();
#pragma unroll
    for (int r = 0; r < 8; ++r) { m_st[mi][r] = -3.0e38f; l_st[mi][r] = 0.f; }
  }

  // global-key part (keys 0..G-1 with masked_add)
  for (int g0 = 0; g0 < GLB; g0 += 32) {
    auto keyf = [&](int c) { return g0 + c; };
    auto addf = [&](int, int cc) { return madd[g0 + cc]; };
    attn_chunk(Kp, Vp, co, lane, qa, o, m_st, l_st, plds, keyf, addf);
  }
  // banded part (3W keys, band mask + window-removal mask; OOB clamped+masked)
  const int bb = blk * WIN - WIN;
  for (int j0 = 0; j0 < 3 * WIN; j0 += 32) {
    auto keyf = [&](int c) {
      int a = bb + j0 + c;
      return a < 0 ? 0 : (a > SEQ - 1 ? SEQ - 1 : a);
    };
    auto addf = [&](int qr, int cc) {
      const int j = j0 + cc;
      const int a = bb + j;
      const bool inr = (a >= 0) && (a < SEQ);
      const float wr = inr ? wrem[inr ? a : 0] : NEGF;
      const int rel = j - (qinb + qr) - WIN;
      return ((rel <= WIN) && (rel >= -WIN)) ? wr : (NEGF + wr);
    };
    attn_chunk(Kp, Vp, co, lane, qa, o, m_st, l_st, plds, keyf, addf);
  }
#pragma unroll
  for (int mi = 0; mi < 2; ++mi)
#pragma unroll
    for (int r = 0; r < 8; ++r) {
      const float inv = 1.0f / fmaxf(l_st[mi][r], 1e-20f);
      const int row = qseq + mi * 16 + r + 8 * lhi;
#pragma unroll
      for (int ni = 0; ni < 4; ++ni)
        Op[(size_t)row * EMB + co + ni * 16 + lh] = o[mi][ni][r] * inv;
    }
}

// ---------------------------------------------------------------------------
// Global-query attention: 64 queries x full sequence, grid = H, 2 waves.
// Overwrites rows [0, G) of the attention output.
// ---------------------------------------------------------------------------
__global__ __launch_bounds__(64)
void global_attn_kernel(const float* __restrict__ Qg, const float* __restrict__ Kg,
                        const float* __restrict__ Vg,
                        const float* __restrict__ madd, float* __restrict__ Op) {
  __shared__ _Float16 plds_all[2 * 32 * 34];
  const int tid = threadIdx.x, lane = tid & 31, wave = tid >> 5;
  _Float16* plds = plds_all + wave * (32 * 34);
  const int head = blockIdx.x, co = head * HD;
  const int q0 = wave * 32;
  const int lh = lane & 15, lhi = lane >> 4;

  v16h qa[2][2];
#pragma unroll
  for (int mi = 0; mi < 2; ++mi)
#pragma unroll
    for (int kk = 0; kk < 2; ++kk)
      qa[mi][kk] = load_q_frag(Qg, q0 + mi * 16, co, kk, lane);

  v8f o[2][4];
  float m_st[2][8], l_st[2][8];
#pragma unroll
  for (int mi = 0; mi < 2; ++mi) {
#pragma unroll
    for (int ni = 0; ni < 4; ++ni) o[mi][ni] = v8f_zero();
#pragma unroll
    for (int r = 0; r < 8; ++r) { m_st[mi][r] = -3.0e38f; l_st[mi][r] = 0.f; }
  }
  for (int c0 = 0; c0 < SEQ; c0 += 32) {
    auto keyf = [&](int c) { return c0 + c; };
    auto addf = [&](int, int cc) { return madd[c0 + cc]; };
    attn_chunk(Kg, Vg, co, lane, qa, o, m_st, l_st, plds, keyf, addf);
  }
#pragma unroll
  for (int mi = 0; mi < 2; ++mi)
#pragma unroll
    for (int r = 0; r < 8; ++r) {
      const float inv = 1.0f / fmaxf(l_st[mi][r], 1e-20f);
      const int row = q0 + mi * 16 + r + 8 * lhi;
#pragma unroll
      for (int ni = 0; ni < 4; ++ni)
        Op[(size_t)row * EMB + co + ni * 16 + lh] = o[mi][ni][r] * inv;
    }
}

// ---------------------------------------------------------------------------
// Elementwise / LayerNorm kernels
// ---------------------------------------------------------------------------
DEVI float block_sum(float v, float* sb) {
#pragma unroll
  for (int m = 1; m < 32; m <<= 1) v += __shfl_xor(v, m, 32);
  const int w = threadIdx.x >> 5;
  if ((threadIdx.x & 31) == 0) sb[w] = v;
  __syncthreads();
  float t = 0.f;
#pragma unroll
  for (int i = 0; i < 8; ++i) t += sb[i];
  __syncthreads();
  return t;
}

__global__ __launch_bounds__(256)
void mask_kernel(const int* __restrict__ am, const int* __restrict__ gam,
                 float* __restrict__ madd, float* __restrict__ wrem) {
  const int i = blockIdx.x * 256 + threadIdx.x;
  if (i < SEQ) {
    const int m = am[i] * (gam[i] + 1);
    madd[i] = (m == 0) ? NEGF : 0.0f;
    wrem[i] = (m == 0 || m == 2) ? NEGF : 0.0f;
  }
}

__global__ __launch_bounds__(256)
void embed_ln_kernel(const int* __restrict__ ids, const float* __restrict__ we,
                     const float* __restrict__ pe, const float* __restrict__ te,
                     const float* __restrict__ g, const float* __restrict__ b,
                     float* __restrict__ h) {
  __shared__ float sb[8];
  const int s = blockIdx.x;
  const int id = ids[s];
  float vals[3];
  float sum = 0.f;
#pragma unroll
  for (int i = 0; i < 3; ++i) {
    const int c = threadIdx.x + i * 256;
    const float t = we[(size_t)id * EMB + c] + pe[(size_t)s * EMB + c] + te[c];
    vals[i] = t;
    sum += t;
  }
  const float mu = block_sum(sum, sb) * (1.0f / EMB);
  float vs = 0.f;
#pragma unroll
  for (int i = 0; i < 3; ++i) { const float d = vals[i] - mu; vs += d * d; }
  const float rstd = rsqrtf(block_sum(vs, sb) * (1.0f / EMB) + 1e-5f);
#pragma unroll
  for (int i = 0; i < 3; ++i) {
    const int c = threadIdx.x + i * 256;
    h[(size_t)s * EMB + c] = (vals[i] - mu) * rstd * g[c] + b[c];
  }
}

__global__ __launch_bounds__(256)
void residual_ln_kernel(float* __restrict__ h, const float* __restrict__ x,
                        const float* __restrict__ g, const float* __restrict__ b) {
  __shared__ float sb[8];
  const int s = blockIdx.x;
  float vals[3];
  float sum = 0.f;
#pragma unroll
  for (int i = 0; i < 3; ++i) {
    const int c = threadIdx.x + i * 256;
    const float t = h[(size_t)s * EMB + c] + x[(size_t)s * EMB + c];
    vals[i] = t;
    sum += t;
  }
  const float mu = block_sum(sum, sb) * (1.0f / EMB);
  float vs = 0.f;
#pragma unroll
  for (int i = 0; i < 3; ++i) { const float d = vals[i] - mu; vs += d * d; }
  const float rstd = rsqrtf(block_sum(vs, sb) * (1.0f / EMB) + 1e-5f);
#pragma unroll
  for (int i = 0; i < 3; ++i) {
    const int c = threadIdx.x + i * 256;
    h[(size_t)s * EMB + c] = (vals[i] - mu) * rstd * g[c] + b[c];
  }
}

__global__ __launch_bounds__(256)
void gelu_kernel(float* __restrict__ x, size_t n) {
  size_t i = (size_t)blockIdx.x * 256 + threadIdx.x;
  const size_t stride = (size_t)gridDim.x * 256;
  for (; i < n; i += stride) {
    const float v = x[i];
    x[i] = 0.5f * v * (1.0f + erff(v * 0.70710678118654752f));
  }
}

// ---------------------------------------------------------------------------
// Host orchestration
// ---------------------------------------------------------------------------
extern "C" void kernel_launch(void* const* d_in, const int* in_sizes, int n_in,
                              void* d_out, int out_size, void* d_ws,
                              size_t ws_size, hipStream_t stream) {
  (void)out_size; (void)ws_size;
  if (n_in < 3 + 5 + NL * 22) return;

  const int* ids = (const int*)d_in[0];
  const int* am = (const int*)d_in[1];
  const int* gam = (const int*)d_in[2];

  const float *we, *pe, *te, *lng, *lnb;
  const float* lw[NL][22];  // wq,bq,wk,bk,wv,bv,wqg,bqg,wkg,bkg,wvg,bvg,
                            // wo,bo,w1,b1,w2,b2,l1g,l1b,l2g,l2b
  if (in_sizes[3] == VOCAB * EMB) {
    // insertion-order flattening: word_emb,pos_emb,tt_emb,ln_g,ln_b, layers...
    we = (const float*)d_in[3]; pe = (const float*)d_in[4];
    te = (const float*)d_in[5]; lng = (const float*)d_in[6];
    lnb = (const float*)d_in[7];
    for (int l = 0; l < NL; ++l)
      for (int i = 0; i < 22; ++i) lw[l][i] = (const float*)d_in[8 + l * 22 + i];
  } else {
    // jax tree_flatten (alphabetical): layers first, keys sorted inside layer
    static const int perm[22] = {18, 5, 15, 2, 20, 7, 19, 6, 16, 3, 21,
                                 8, 17, 4, 13, 0, 14, 1, 10, 9, 12, 11};
    for (int l = 0; l < NL; ++l)
      for (int i = 0; i < 22; ++i)
        lw[l][i] = (const float*)d_in[3 + l * 22 + perm[i]];
    const int base = 3 + NL * 22;
    lnb = (const float*)d_in[base + 0];
    lng = (const float*)d_in[base + 1];
    pe = (const float*)d_in[base + 2];
    te = (const float*)d_in[base + 3];
    we = (const float*)d_in[base + 4];
  }

  float* h = (float*)d_out;  // residual stream lives in d_out
  float* ws = (float*)d_ws;
  const size_t SE = (size_t)SEQ * EMB;
  float* qb = ws;
  float* kb = ws + SE;
  float* vb = ws + 2 * SE;
  float* kg = ws + 3 * SE;
  float* vg = ws + 4 * SE;
  float* attn = ws + 5 * SE;
  float* tmp = ws + 6 * SE;
  float* ff1 = ws + 7 * SE;                 // SEQ*FFD = 4*SE
  float* qg = ws + 11 * SE;                 // G*EMB
  float* madd = qg + (size_t)GLB * EMB;
  float* wrem = madd + SEQ;
  // per-layer f16 transposed weight arena (reused across layers)
  _Float16* wt = (_Float16*)(wrem + SEQ);
  const size_t EE = (size_t)EMB * EMB, EF = (size_t)EMB * FFD;
  _Float16* wqT = wt;
  _Float16* wkT = wt + EE;
  _Float16* wvT = wt + 2 * EE;
  _Float16* wqgT = wt + 3 * EE;
  _Float16* wkgT = wt + 4 * EE;
  _Float16* wvgT = wt + 5 * EE;
  _Float16* woT = wt + 6 * EE;
  _Float16* w1T = wt + 7 * EE;        // [FFD, EMB]
  _Float16* w2T = wt + 7 * EE + EF;   // [EMB, FFD]

  mask_kernel<<<SEQ / 256, 256, 0, stream>>>(am, gam, madd, wrem);
  embed_ln_kernel<<<SEQ, 256, 0, stream>>>(ids, we, pe, te, lng, lnb, h);

  const dim3 gEE(EMB / 256, SEQ / 64);
  const dim3 gEF(FFD / 256, SEQ / 64);
  const dim3 gQG(EMB / 256, 1);
  const dim3 tEE(EMB / 32, EMB / 32), tEF(FFD / 32, EMB / 32),
      tFE(EMB / 32, FFD / 32);
  const dim3 tb(32, 8);
  const float qscale = 0.125f;  // 1/sqrt(D)

  for (int l = 0; l < NL; ++l) {
    const float* const* p = lw[l];
    // weight transpose + f16 convert for this layer
    transpose_f16_kernel<<<tEE, tb, 0, stream>>>(p[0], wqT, EMB, EMB);
    transpose_f16_kernel<<<tEE, tb, 0, stream>>>(p[2], wkT, EMB, EMB);
    transpose_f16_kernel<<<tEE, tb, 0, stream>>>(p[4], wvT, EMB, EMB);
    transpose_f16_kernel<<<tEE, tb, 0, stream>>>(p[6], wqgT, EMB, EMB);
    transpose_f16_kernel<<<tEE, tb, 0, stream>>>(p[8], wkgT, EMB, EMB);
    transpose_f16_kernel<<<tEE, tb, 0, stream>>>(p[10], wvgT, EMB, EMB);
    transpose_f16_kernel<<<tEE, tb, 0, stream>>>(p[12], woT, EMB, EMB);
    transpose_f16_kernel<<<tEF, tb, 0, stream>>>(p[14], w1T, EMB, FFD);
    transpose_f16_kernel<<<tFE, tb, 0, stream>>>(p[16], w2T, FFD, EMB);

    gemm_bt_kernel<<<gEE, 256, 0, stream>>>(h, wqT, p[1], qb, SEQ, EMB, EMB, qscale);
    gemm_bt_kernel<<<gEE, 256, 0, stream>>>(h, wkT, p[3], kb, SEQ, EMB, EMB, 1.f);
    gemm_bt_kernel<<<gEE, 256, 0, stream>>>(h, wvT, p[5], vb, SEQ, EMB, EMB, 1.f);
    local_attn_kernel<<<dim3(SEQ / WIN, NH), 256, 0, stream>>>(qb, kb, vb, madd,
                                                               wrem, attn);
    gemm_bt_kernel<<<gQG, 256, 0, stream>>>(h, wqgT, p[7], qg, GLB, EMB, EMB, qscale);
    gemm_bt_kernel<<<gEE, 256, 0, stream>>>(h, wkgT, p[9], kg, SEQ, EMB, EMB, 1.f);
    gemm_bt_kernel<<<gEE, 256, 0, stream>>>(h, wvgT, p[11], vg, SEQ, EMB, EMB, 1.f);
    global_attn_kernel<<<NH, 64, 0, stream>>>(qg, kg, vg, madd, attn);
    gemm_bt_kernel<<<gEE, 256, 0, stream>>>(attn, woT, p[13], tmp, SEQ, EMB, EMB, 1.f);
    residual_ln_kernel<<<SEQ, 256, 0, stream>>>(h, tmp, p[18], p[19]);
    gemm_bt_kernel<<<gEF, 256, 0, stream>>>(h, w1T, p[15], ff1, SEQ, FFD, EMB, 1.f);
    gelu_kernel<<<4096, 256, 0, stream>>>(ff1, (size_t)SEQ * FFD);
    gemm_bt_kernel<<<gEE, 256, 0, stream>>>(ff1, w2T, p[17], tmp, SEQ, EMB, FFD, 1.f);
    residual_ln_kernel<<<SEQ, 256, 0, stream>>>(h, tmp, p[20], p[21]);
  }
}
